// SkeletonAwareSpatialAttention_7138235646317
// MI455X (gfx1250) — matrix operand order repaired and checked by reading, not compiled, not run on hardware
//
#include <hip/hip_runtime.h>
#include <hip/hip_bf16.h>

typedef __attribute__((ext_vector_type(16))) _Float16 v16h;
typedef __attribute__((ext_vector_type(8)))  float    v8f;

#define NTOK    (64*512)
#define DM      128
#define NH      8
#define WAVES   4
#define BLOCKS  2048

// ---- LDS layout (byte offsets into dynamic smem) ----
#define OFF_WQB    0           // 128 * float4  (W0,W1,W2,bias) for Q
#define OFF_WKB    2048
#define OFF_WVB    4096
#define OFF_WOT    6144        // Wo^T as f16, 128 rows * 272B (padded, conflict-free)
#define WOT_STRIDE 272
#define OFF_BO     40960       // 128 floats
#define OFF_WAVE   41472       // per-wave region
#define PWS        15872
#define PW_XS      0           // 16 * float4 {x,y,z,0}
#define PW_KS      256         // k row-major f16, 16 rows * 272B
#define KS_STRIDE  272
#define PW_VT      4608        // v transposed f16, 128 rows * 48B
#define VT_STRIDE  48
#define PW_WS      10752       // softmax weights f16, 16 rows * 48B
#define WS_STRIDE  48
#define PW_CS      11520       // ctx f16 row-major, 16 rows * 272B
#define CS_STRIDE  272
#define SMEM_BYTES (OFF_WAVE + WAVES*PWS)   // 104960

// Kinematic adjacency as bitmasks: adj(m,n) = (m==n) + skel + 0.8*sym
__device__ const unsigned short SKEL_MASK[16] = {
  0x2492, 0x0005, 0x000A, 0x0004, 0x0021, 0x0050, 0x0020, 0x0101,
  0x0280, 0x0100, 0x0801, 0x1400, 0x0800, 0x4001, 0xA000, 0x4000 };
__device__ const unsigned short SYM_MASK[16] = {
  0x0000, 0x0000, 0x0000, 0x0000, 0x0080, 0x0100, 0x0200, 0x0010,
  0x0020, 0x0040, 0x2000, 0x4000, 0x8000, 0x0400, 0x0800, 0x1000 };

union U16 { v16h v; uint4 u[2]; };

static __device__ __forceinline__ unsigned pack2(float a, float b) {
  union { _Float16 h[2]; unsigned u; } t;
  t.h[0] = (_Float16)a; t.h[1] = (_Float16)b;
  return t.u;
}

__global__ void __launch_bounds__(WAVES*32)
sasa_kernel(const float* __restrict__ x,
            const float* __restrict__ Wq, const float* __restrict__ bq,
            const float* __restrict__ Wk, const float* __restrict__ bk,
            const float* __restrict__ Wv, const float* __restrict__ bv,
            const float* __restrict__ Wo, const float* __restrict__ bo,
            float* __restrict__ out)
{
  extern __shared__ char smem[];
  float4* wqb  = (float4*)(smem + OFF_WQB);
  float4* wkb  = (float4*)(smem + OFF_WKB);
  float4* wvb  = (float4*)(smem + OFF_WVB);
  char*   wot  = smem + OFF_WOT;
  float*  bo_s = (float*)(smem + OFF_BO);

  const int tid = threadIdx.x;

  // ---- once-per-block staging (blockDim == 128) ----
  {
    int d = tid;
    wqb[d]  = make_float4(Wq[d], Wq[128+d], Wq[256+d], bq[d]);
    wkb[d]  = make_float4(Wk[d], Wk[128+d], Wk[256+d], bk[d]);
    wvb[d]  = make_float4(Wv[d], Wv[128+d], Wv[256+d], bv[d]);
    bo_s[d] = bo[d];
    for (int idx = tid; idx < 128*128; idx += 128) {
      int k = idx >> 7, n = idx & 127;
      *(_Float16*)(wot + n*WOT_STRIDE + k*2) = (_Float16)Wo[idx];  // Wo^T
    }
  }
  __syncthreads();

  const int wave = tid >> 5;
  const int lane = tid & 31;
  const int jn   = lane & 15;   // A-row j / B-col N / C-col N
  const int half = lane >> 4;

  char*   pw = smem + OFF_WAVE + wave*PWS;
  float4* xs = (float4*)(pw + PW_XS);
  char*   ks = pw + PW_KS;
  char*   vt = pw + PW_VT;
  char*   ws = pw + PW_WS;
  char*   cs = pw + PW_CS;

  // adjacency bias for this lane's C positions (M = r + 8*half, N = jn)
  float aj[8];
  #pragma unroll
  for (int r = 0; r < 8; ++r) {
    int M = r + 8*half;
    float a = (M == jn) ? 1.0f : 0.0f;
    a += (float)((SKEL_MASK[M] >> jn) & 1);
    a += 0.8f * (float)((SYM_MASK[M] >> jn) & 1);
    aj[r] = a;
  }

  uint4 qpk[NH];  // per-head q A-fragment (8 f16, pad K=16..31 with zeros at use)
  const uint4 z4 = make_uint4(0u, 0u, 0u, 0u);

  const int gwave = blockIdx.x * WAVES + wave;
  const int step  = gridDim.x * WAVES;

  for (int tok = gwave; tok < NTOK; tok += step) {
    // ---- stage this token's joints ----
    if (lane < 16) {
      const float* xp = x + (size_t)tok*48 + lane*3;
      xs[lane] = make_float4(xp[0], xp[1], xp[2], 0.0f);
    }
    asm volatile("" ::: "memory");   // keep LDS program order (DS is in-order in-wave)

    // ---- q,k projection: lane owns row jn, cols d%16 in [8*half, 8*half+8) ----
    float4 xv = xs[jn];
    #pragma unroll
    for (int h = 0; h < NH; ++h) {
      float qv[8], kv[8];
      #pragma unroll
      for (int e = 0; e < 8; ++e) {
        int d = h*16 + half*8 + e;
        float4 wq = wqb[d], wk = wkb[d];
        qv[e] = wq.w + xv.x*wq.x + xv.y*wq.y + xv.z*wq.z;
        kv[e] = wk.w + xv.x*wk.x + xv.y*wk.y + xv.z*wk.z;
      }
      qpk[h] = make_uint4(pack2(qv[0],qv[1]), pack2(qv[2],qv[3]),
                          pack2(qv[4],qv[5]), pack2(qv[6],qv[7]));
      *(uint4*)(ks + jn*KS_STRIDE + h*32 + half*16) =
          make_uint4(pack2(kv[0],kv[1]), pack2(kv[2],kv[3]),
                     pack2(kv[4],kv[5]), pack2(kv[6],kv[7]));
    }
    // ---- v projection, stored transposed: vt[d][j] ----
    #pragma unroll
    for (int m = 0; m < 4; ++m) {
      int d = m*32 + lane;
      float4 wv = wvb[d];
      float vv[16];
      #pragma unroll
      for (int jj = 0; jj < 16; ++jj) {
        float4 xq = xs[jj];
        vv[jj] = wv.w + xq.x*wv.x + xq.y*wv.y + xq.z*wv.z;
      }
      *(uint4*)(vt + d*VT_STRIDE) =
          make_uint4(pack2(vv[0],vv[1]), pack2(vv[2],vv[3]),
                     pack2(vv[4],vv[5]), pack2(vv[6],vv[7]));
      *(uint4*)(vt + d*VT_STRIDE + 16) =
          make_uint4(pack2(vv[8],vv[9]), pack2(vv[10],vv[11]),
                     pack2(vv[12],vv[13]), pack2(vv[14],vv[15]));
    }
    asm volatile("" ::: "memory");

    // ---- attention, one 16x16x32 WMMA pair per head (K padded 16->32) ----
    #pragma unroll
    for (int h = 0; h < NH; ++h) {
      U16 a; a.u[0] = qpk[h]; a.u[1] = z4;
      U16 b; b.u[0] = *(const uint4*)(ks + jn*KS_STRIDE + h*32 + half*16); b.u[1] = z4;
      v8f c = {};
      c = __builtin_amdgcn_wmma_f32_16x16x32_f16(false, a.v, false, b.v,
                                                 (short)0, c, false, false);
      // (scores + adj) / 4, then row softmax across N (16 lanes of this half)
      #pragma unroll
      for (int r = 0; r < 8; ++r) {
        float s = (c[r] + aj[r]) * 0.25f;
        float mx = s;
        mx = fmaxf(mx, __shfl_xor(mx, 1));
        mx = fmaxf(mx, __shfl_xor(mx, 2));
        mx = fmaxf(mx, __shfl_xor(mx, 4));
        mx = fmaxf(mx, __shfl_xor(mx, 8));
        float e = __expf(s - mx);
        float sum = e;
        sum += __shfl_xor(sum, 1);
        sum += __shfl_xor(sum, 2);
        sum += __shfl_xor(sum, 4);
        sum += __shfl_xor(sum, 8);
        float w = e * __frcp_rn(sum);
        *(_Float16*)(ws + (r + 8*half)*WS_STRIDE + jn*2) = (_Float16)w;
      }
      asm volatile("" ::: "memory");
      // ctx = softmax(w) @ v
      U16 aw; aw.u[0] = *(const uint4*)(ws + jn*WS_STRIDE + half*16); aw.u[1] = z4;
      U16 bv2; bv2.u[0] = *(const uint4*)(vt + (h*16 + jn)*VT_STRIDE + half*16); bv2.u[1] = z4;
      v8f cc = {};
      cc = __builtin_amdgcn_wmma_f32_16x16x32_f16(false, aw.v, false, bv2.v,
                                                  (short)0, cc, false, false);
      #pragma unroll
      for (int r = 0; r < 8; ++r)
        *(_Float16*)(cs + (r + 8*half)*CS_STRIDE + (h*16 + jn)*2) = (_Float16)cc[r];
      asm volatile("" ::: "memory");
    }

    // ---- output projection: ctx[16x128] @ Wo[128x128], 8 tiles x 4 K-chunks ----
    // A-fragments depend only on the K-chunk: load once, reuse across all 8 tiles.
    U16 af[4];
    #pragma unroll
    for (int ch = 0; ch < 4; ++ch) {
      af[ch].u[0] = *(const uint4*)(cs + jn*CS_STRIDE + (ch*32 + half*8)*2);
      af[ch].u[1] = *(const uint4*)(cs + jn*CS_STRIDE + (ch*32 + 16 + half*8)*2);
    }
    float* outp = out + (size_t)tok * (16*DM);
    #pragma unroll
    for (int t = 0; t < 8; ++t) {
      v8f o = {};
      #pragma unroll
      for (int ch = 0; ch < 4; ++ch) {
        U16 b;
        b.u[0] = *(const uint4*)(wot + (t*16 + jn)*WOT_STRIDE + (ch*32 + half*8)*2);
        b.u[1] = *(const uint4*)(wot + (t*16 + jn)*WOT_STRIDE + (ch*32 + 16 + half*8)*2);
        o = __builtin_amdgcn_wmma_f32_16x16x32_f16(false, af[ch].v, false, b.v,
                                                   (short)0, o, false, false);
      }
      float bb = bo_s[t*16 + jn];
      #pragma unroll
      for (int r = 0; r < 8; ++r)
        outp[(size_t)(r + 8*half)*DM + t*16 + jn] = o[r] + bb;
    }
  }
}

extern "C" void kernel_launch(void* const* d_in, const int* in_sizes, int n_in,
                              void* d_out, int out_size, void* d_ws, size_t ws_size,
                              hipStream_t stream) {
  (void)in_sizes; (void)n_in; (void)out_size; (void)d_ws; (void)ws_size;
  const float* x  = (const float*)d_in[0];
  const float* Wq = (const float*)d_in[1];
  const float* bq = (const float*)d_in[2];
  const float* Wk = (const float*)d_in[3];
  const float* bk = (const float*)d_in[4];
  const float* Wv = (const float*)d_in[5];
  const float* bv = (const float*)d_in[6];
  const float* Wo = (const float*)d_in[7];
  const float* bo = (const float*)d_in[8];
  float* out = (float*)d_out;
  sasa_kernel<<<dim3(BLOCKS), dim3(WAVES*32), SMEM_BYTES, stream>>>(
      x, Wq, bq, Wk, bk, Wv, bv, Wo, bo, out);
}
